// SelfAttentionLayer_52802327937453
// MI455X (gfx1250) — matrix operand order repaired and checked by reading, not compiled, not run on hardware
//
#include <hip/hip_runtime.h>
#include <hip/hip_bf16.h>
#include <stdint.h>

#define B_  8
#define S_  1024
#define H_  768
#define NH_ 12
#define DH_ 64
#define M_  (B_ * S_)

typedef __attribute__((ext_vector_type(16))) __bf16 v16bf;
typedef __attribute__((ext_vector_type(8)))  float  v8f;

union Frag16 {
  v16bf v;
  unsigned int u[8];
  uint4 q[2];
};
union AccF {
  v8f v;
  float f[8];
};

__device__ __forceinline__ unsigned short f2bf(float x) {
  unsigned int u = __float_as_uint(x);
  u += 0x7fffu + ((u >> 16) & 1u);           // round-to-nearest-even
  return (unsigned short)(u >> 16);
}

__device__ __forceinline__ v8f wmma_bf16(const Frag16& a, const Frag16& b, v8f c) {
  // D = A(16x32 bf16) x B(32x16 bf16) + C(16x16 f32)
  return __builtin_amdgcn_wmma_f32_16x16x32_bf16(
      /*neg_a=*/false, a.v, /*neg_b=*/false, b.v,
      /*c_mod=*/(short)0, c, /*reuse_a=*/false, /*reuse_b=*/false);
}

// ---------------------------------------------------------------------------
// Tiled GEMM: out[MxN] = A[MxK] (f32 or bf16) @ W[KxN] (f32->bf16) + bias
// Block tile 128x128, K-step 32, 8 waves (2x4), each wave 64x32 = 8 WMMA tiles.
// LDS tiles padded to 20-dword rows: keeps ds_load_b128 16B alignment while
// spreading scattered stores / strided fragment reads across banks.
// TRANSV: emit bf16 output in [b][h][d][s] layout (for attention V operand).
// ---------------------------------------------------------------------------
template<bool A_F32, bool RELU, bool TRANSV>
__global__ __launch_bounds__(256) void gemm_bf16_kernel(
    const void* __restrict__ Ap, const float* __restrict__ Wt,
    const float* __restrict__ bias,
    float* __restrict__ outf, unsigned short* __restrict__ outbf,
    int M, int N, int K)
{
  __shared__ __align__(16) unsigned int Asub[128][20];   // [m][k-pair] bf16x2
  __shared__ __align__(16) unsigned int Btsub[128][20];  // [n][k-pair] bf16x2

  const int tid  = threadIdx.x;
  const int wid  = tid >> 5;
  const int lane = tid & 31;
  const int lo   = lane & 15;
  const int hi   = lane >> 4;
  const int wm   = wid >> 2;      // 0..1
  const int wn   = wid & 3;       // 0..3
  const int m0   = blockIdx.y * 128;
  const int n0   = blockIdx.x * 128;

  const float*          Af = (const float*)Ap;
  const unsigned short* Ab = (const unsigned short*)Ap;

  AccF acc[4][2];
  #pragma unroll
  for (int mt = 0; mt < 4; ++mt)
    #pragma unroll
    for (int nt = 0; nt < 2; ++nt)
      #pragma unroll
      for (int v = 0; v < 8; ++v) acc[mt][nt].f[v] = 0.0f;

  for (int k0 = 0; k0 < K; k0 += 32) {
    __syncthreads();
    // ---- A tile: coalesced (float2 / dword per lane along k)
    #pragma unroll
    for (int i = 0; i < 8; ++i) {
      int p  = tid + i * 256;
      int m  = p >> 4;          // 0..127
      int kp = p & 15;          // k-pair 0..15
      unsigned int pk;
      if (A_F32) {
        const float2 f = *(const float2*)(Af + (size_t)(m0 + m) * K + k0 + kp * 2);
        pk = (unsigned)f2bf(f.x) | ((unsigned)f2bf(f.y) << 16);
      } else {
        pk = *(const unsigned int*)(Ab + (size_t)(m0 + m) * K + k0 + kp * 2);
      }
      Asub[m][kp] = pk;
    }
    // ---- B tile: coalesced along n (contiguous 512B row segments), k-pairs
    //      packed into LDS [n][kp]
    #pragma unroll
    for (int i = 0; i < 8; ++i) {
      int idx = tid + i * 256;  // 0..2047
      int n   = idx & 127;
      int k2  = idx >> 7;       // k-pair 0..15
      const float* ws = Wt + (size_t)(k0 + 2 * k2) * N + n0 + n;
      Btsub[n][k2] = (unsigned)f2bf(ws[0]) | ((unsigned)f2bf(ws[N]) << 16);
    }
    __syncthreads();

    if (k0 + 32 < K) {   // prefetch next K tile (global_prefetch_b8)
      int m = tid >> 4, kp = tid & 15;
      if (A_F32) __builtin_prefetch(Af + (size_t)(m0 + m) * K + k0 + 32 + kp * 2, 0, 0);
      else       __builtin_prefetch(Ab + (size_t)(m0 + m) * K + k0 + 32 + kp * 2, 0, 0);
      __builtin_prefetch(Wt + (size_t)(k0 + 32 + 2 * (tid >> 7)) * N + n0 + (tid & 127), 0, 0);
    }

    // A fragment (ISA 16-bit A 16x32 layout): lane = row (lo), hi selects K half
    Frag16 afrag[4];
    #pragma unroll
    for (int mt = 0; mt < 4; ++mt) {
      int row = wm * 64 + mt * 16 + lo;
      afrag[mt].q[0] = *(const uint4*)&Asub[row][hi * 4];       // K = hi*8 + 0..7
      afrag[mt].q[1] = *(const uint4*)&Asub[row][8 + hi * 4];   // K = 16+hi*8+0..7
    }
    #pragma unroll
    for (int nt = 0; nt < 2; ++nt) {
      // B fragment (32x16): lane = col (lo), K = hi*8 + 0..7 and 16+hi*8+0..7
      Frag16 bfrag;
      int n = wn * 32 + nt * 16 + lo;
      bfrag.q[0] = *(const uint4*)&Btsub[n][hi * 4];
      bfrag.q[1] = *(const uint4*)&Btsub[n][8 + hi * 4];
      #pragma unroll
      for (int mt = 0; mt < 4; ++mt)
        acc[mt][nt].v = wmma_bf16(afrag[mt], bfrag, acc[mt][nt].v);
    }
  }

  #pragma unroll
  for (int nt = 0; nt < 2; ++nt) {
    int n = n0 + wn * 32 + nt * 16 + lo;
    float bv = bias[n];
    #pragma unroll
    for (int mt = 0; mt < 4; ++mt) {
      #pragma unroll
      for (int v = 0; v < 8; ++v) {
        int m = m0 + wm * 64 + mt * 16 + v + 8 * hi;   // C/D layout: M=v+8*hi
        float y = acc[mt][nt].f[v] + bv;
        if (RELU) y = fmaxf(y, 0.0f);
        if (outf)  outf[(size_t)m * N + n] = y;
        if (outbf) {
          if (TRANSV) {
            // m = b*S + s ; n = h*DH + d  ->  vt[((b*NH + h)*DH + d)*S + s]
            int bb = m >> 10, ss = m & (S_ - 1);
            int hh = n >> 6,  dd = n & (DH_ - 1);
            outbf[((size_t)((bb * NH_ + hh) * DH_ + dd) << 10) + ss] = f2bf(y);
          } else {
            outbf[(size_t)m * N + n] = f2bf(y);
          }
        }
      }
    }
  }
}

// ---------------------------------------------------------------------------
// Flash-style attention, post-softmax mask: ctx = (softmax(QK^T/8)*mask) @ V
// One wave per (b, h, 16-row Q tile); 32 KV columns per step.
// Softmax denominator uses UNMASKED exp; PV accumulation uses exp*mask.
// V comes pre-transposed as vt[b][h][d][s], so all fragments are b128 loads.
// ---------------------------------------------------------------------------
__global__ __launch_bounds__(128) void attn_kernel(
    const unsigned short* __restrict__ qb,
    const unsigned short* __restrict__ kb,
    const unsigned short* __restrict__ vt,
    const float* __restrict__ mask,
    float* __restrict__ ctx)
{
  __shared__ __align__(16) unsigned int plds[4][16][16];   // per-wave P tile

  const int tid  = threadIdx.x;
  const int wid  = tid >> 5;
  const int lane = tid & 31;
  const int lo   = lane & 15;
  const int hi   = lane >> 4;

  const int gid  = blockIdx.x * 4 + wid;   // 0..6143
  const int qt   = gid & 63;               // q tile (S/16)
  const int rest = gid >> 6;
  const int h    = rest % NH_;
  const int b    = rest / NH_;

  // Q fragments (A layout), K = DH = 64 -> two 16x32 fragments
  Frag16 aq[2];
  {
    const unsigned short* qbase = qb + ((size_t)(b * S_ + qt * 16 + lo)) * H_ + h * DH_;
    #pragma unroll
    for (int f = 0; f < 2; ++f) {
      aq[f].q[0] = *(const uint4*)(qbase + f * 32 + hi * 8);
      aq[f].q[1] = *(const uint4*)(qbase + f * 32 + 16 + hi * 8);
    }
  }

  float mrow[8], srow[8];
  AccF o[4];
  #pragma unroll
  for (int v = 0; v < 8; ++v) { mrow[v] = -3.0e38f; srow[v] = 0.0f; }
  #pragma unroll
  for (int nt = 0; nt < 4; ++nt)
    #pragma unroll
    for (int v = 0; v < 8; ++v) o[nt].f[v] = 0.0f;

  const float* mbase = mask + (size_t)b * S_ * S_;

  for (int kv0 = 0; kv0 < S_; kv0 += 32) {
    // ---- scores: Q(16x64) x K^T(64x32) -> two 16x16 tiles
    AccF c[2];
    #pragma unroll
    for (int t = 0; t < 2; ++t) {
      const unsigned short* kbase =
          kb + ((size_t)(b * S_ + kv0 + t * 16 + lo)) * H_ + h * DH_;
      Frag16 kf0, kf1;                       // B layout: col=lo, K pairs in VGPRs
      kf0.q[0] = *(const uint4*)(kbase + hi * 8);
      kf0.q[1] = *(const uint4*)(kbase + 16 + hi * 8);
      kf1.q[0] = *(const uint4*)(kbase + 32 + hi * 8);
      kf1.q[1] = *(const uint4*)(kbase + 32 + 16 + hi * 8);
      v8f z = {0, 0, 0, 0, 0, 0, 0, 0};
      c[t].v = wmma_bf16(aq[0], kf0, z);
      c[t].v = wmma_bf16(aq[1], kf1, c[t].v);
      #pragma unroll
      for (int v = 0; v < 8; ++v) c[t].f[v] *= 0.125f;   // 1/sqrt(64)
    }

    // ---- online softmax stats (row = v + 8*hi, spread over 16 lanes)
    float tm[8], alpha[8], p0[8], p1[8], ssum[8];
    #pragma unroll
    for (int v = 0; v < 8; ++v) tm[v] = fmaxf(c[0].f[v], c[1].f[v]);
    #pragma unroll
    for (int off = 8; off >= 1; off >>= 1)
      #pragma unroll
      for (int v = 0; v < 8; ++v)
        tm[v] = fmaxf(tm[v], __shfl_xor(tm[v], off, 32));
    #pragma unroll
    for (int v = 0; v < 8; ++v) {
      float mnew = fmaxf(mrow[v], tm[v]);
      alpha[v] = __expf(mrow[v] - mnew);
      mrow[v]  = mnew;
      p0[v] = __expf(c[0].f[v] - mnew);
      p1[v] = __expf(c[1].f[v] - mnew);
      ssum[v] = p0[v] + p1[v];
    }
    #pragma unroll
    for (int off = 8; off >= 1; off >>= 1)
      #pragma unroll
      for (int v = 0; v < 8; ++v)
        ssum[v] += __shfl_xor(ssum[v], off, 32);
    #pragma unroll
    for (int v = 0; v < 8; ++v) srow[v] = srow[v] * alpha[v] + ssum[v];

    // ---- post-softmax mask on the PV operand only
    #pragma unroll
    for (int v = 0; v < 8; ++v) {
      const int r = qt * 16 + v + 8 * hi;
      p0[v] *= mbase[(size_t)r * S_ + kv0 + lo];
      p1[v] *= mbase[(size_t)r * S_ + kv0 + 16 + lo];
    }
    #pragma unroll
    for (int nt = 0; nt < 4; ++nt)
      #pragma unroll
      for (int v = 0; v < 8; ++v) o[nt].f[v] *= alpha[v];

    // ---- C-layout -> A-layout transpose through LDS
    __syncthreads();
    {
      unsigned short* pl = (unsigned short*)&plds[wid][0][0];
      #pragma unroll
      for (int v = 0; v < 8; ++v) {
        const int r = v + 8 * hi;
        pl[r * 32 + lo]      = f2bf(p0[v]);
        pl[r * 32 + 16 + lo] = f2bf(p1[v]);
      }
    }
    __syncthreads();
    Frag16 pa;
    pa.q[0] = *(const uint4*)&plds[wid][lo][hi * 4];
    pa.q[1] = *(const uint4*)&plds[wid][lo][8 + hi * 4];

    // ---- PV: P(16x32) x V(32x64); V fragments are b128 loads from vt[b][h][d][s]
    #pragma unroll
    for (int nt = 0; nt < 4; ++nt) {
      Frag16 vf;
      const unsigned short* vtb =
          vt + ((size_t)((b * NH_ + h) * DH_ + nt * 16 + lo)) * S_ + kv0 + hi * 8;
      vf.q[0] = *(const uint4*)(vtb);
      vf.q[1] = *(const uint4*)(vtb + 16);
      o[nt].v = wmma_bf16(pa, vf, o[nt].v);
    }
  }

  #pragma unroll
  for (int v = 0; v < 8; ++v) srow[v] = 1.0f / srow[v];
  #pragma unroll
  for (int nt = 0; nt < 4; ++nt)
    #pragma unroll
    for (int v = 0; v < 8; ++v) {
      const size_t m = (size_t)(b * S_ + qt * 16 + v + 8 * hi);
      ctx[m * H_ + h * DH_ + nt * 16 + lo] = o[nt].f[v] * srow[v];
    }
}

// ---------------------------------------------------------------------------
// Residual + LayerNorm; optionally also emit a bf16 copy for the next GEMM.
// ---------------------------------------------------------------------------
template<bool WBF16>
__global__ __launch_bounds__(256) void ln_kernel(
    const float* __restrict__ a, const float* __restrict__ res,
    const float* __restrict__ gamma, const float* __restrict__ beta,
    float* __restrict__ outf, unsigned short* __restrict__ outbf, int W)
{
  __shared__ float s1[256];
  __shared__ float s2[256];
  const int row = blockIdx.x;
  const int tid = threadIdx.x;
  const float* pa = a + (size_t)row * W;
  const float* pr = res + (size_t)row * W;
  float vals[4];
  float lsum = 0.f, lsq = 0.f;
  int j = 0;
  for (int i = tid; i < W; i += 256, ++j) {
    float v = pa[i] + pr[i];
    vals[j] = v;
    lsum += v; lsq += v * v;
  }
  s1[tid] = lsum; s2[tid] = lsq;
  __syncthreads();
  for (int off = 128; off > 0; off >>= 1) {
    if (tid < off) { s1[tid] += s1[tid + off]; s2[tid] += s2[tid + off]; }
    __syncthreads();
  }
  const float mu   = s1[0] / W;
  const float var  = s2[0] / W - mu * mu;
  const float rstd = rsqrtf(var + 1e-5f);
  j = 0;
  for (int i = tid; i < W; i += 256, ++j) {
    float y = (vals[j] - mu) * rstd * gamma[i] + beta[i];
    outf[(size_t)row * W + i] = y;
    if (WBF16) outbf[(size_t)row * W + i] = f2bf(y);
  }
}

// ---------------------------------------------------------------------------
extern "C" void kernel_launch(void* const* d_in, const int* in_sizes, int n_in,
                              void* d_out, int out_size, void* d_ws, size_t ws_size,
                              hipStream_t stream) {
  (void)in_sizes; (void)n_in; (void)out_size; (void)ws_size;
  const float* x1   = (const float*)d_in[0];
  const float* mask = (const float*)d_in[1];
  const float* Wq   = (const float*)d_in[2];
  const float* bq   = (const float*)d_in[3];
  const float* Wk   = (const float*)d_in[4];
  const float* bk   = (const float*)d_in[5];
  const float* Wv   = (const float*)d_in[6];
  const float* bv   = (const float*)d_in[7];
  const float* g1   = (const float*)d_in[8];
  const float* be1  = (const float*)d_in[9];
  const float* g2   = (const float*)d_in[10];
  const float* be2  = (const float*)d_in[11];
  const float* W1   = (const float*)d_in[12];
  const float* b1   = (const float*)d_in[13];
  const float* W2   = (const float*)d_in[14];
  const float* b2   = (const float*)d_in[15];

  // workspace layout (~168 MB)
  char* w = (char*)d_ws;
  float* q_f = (float*)w;                     w += (size_t)M_ * H_ * 4;
  unsigned short* q_b  = (unsigned short*)w;  w += (size_t)M_ * H_ * 2;
  unsigned short* k_b  = (unsigned short*)w;  w += (size_t)M_ * H_ * 2;
  unsigned short* vt_b = (unsigned short*)w;  w += (size_t)M_ * H_ * 2;   // [b][h][d][s]
  float* ctx = (float*)w;                     w += (size_t)M_ * H_ * 4;
  float* x_f = (float*)w;                     w += (size_t)M_ * H_ * 4;
  unsigned short* x_b  = (unsigned short*)w;  w += (size_t)M_ * H_ * 2;
  unsigned short* h1_b = (unsigned short*)w;  w += (size_t)M_ * (2 * H_) * 2;
  float* ff  = (float*)w;                     w += (size_t)M_ * H_ * 4;

  dim3 blk(256);
  dim3 g_qkv(H_ / 128, M_ / 128);                  // (6, 64)

  // QKV projections (q also fp32 for the residual path; v written transposed)
  gemm_bf16_kernel<true,  false, false><<<g_qkv, blk, 0, stream>>>(x1, Wq, bq, q_f,    q_b,  M_, H_, H_);
  gemm_bf16_kernel<true,  false, false><<<g_qkv, blk, 0, stream>>>(x1, Wk, bk, nullptr, k_b, M_, H_, H_);
  gemm_bf16_kernel<true,  false, true ><<<g_qkv, blk, 0, stream>>>(x1, Wv, bv, nullptr, vt_b, M_, H_, H_);

  // flash attention: 6144 (b,h,qtile) waves, 4 per block
  attn_kernel<<<dim3((B_ * NH_ * (S_ / 16)) / 4), dim3(128), 0, stream>>>(q_b, k_b, vt_b, mask, ctx);

  // x = LN(ctx + q_mixed)
  ln_kernel<true><<<dim3(M_), blk, 0, stream>>>(ctx, q_f, g1, be1, x_f, x_b, H_);

  // FFN
  gemm_bf16_kernel<false, true,  false><<<dim3((2 * H_) / 128, M_ / 128), blk, 0, stream>>>(x_b,  W1, b1, nullptr, h1_b, M_, 2 * H_, H_);
  gemm_bf16_kernel<false, false, false><<<g_qkv, blk, 0, stream>>>(h1_b, W2, b2, ff, nullptr, M_, H_, 2 * H_);

  // out = LN(x + ff)
  ln_kernel<false><<<dim3(M_), blk, 0, stream>>>(ff, x_f, g2, be2, (float*)d_out, nullptr, H_);
}
